// CausalAttention_77146202571221
// MI455X (gfx1250) — matrix operand order repaired and checked by reading, not compiled, not run on hardware
//
#include <hip/hip_runtime.h>
#include <hip/hip_bf16.h>

typedef __attribute__((ext_vector_type(16))) __bf16 v16bf;
typedef __attribute__((ext_vector_type(8)))  __bf16 v8bf;
typedef __attribute__((ext_vector_type(8)))  float  v8f;

#define BATCH 4
#define NSEQ  1024
#define HDIM  1024
#define NHEAD 16
#define HD    64
#define MTOT  (BATCH * NSEQ)   // 4096

static __device__ __forceinline__ v16bf cat8(v8bf lo, v8bf hi) {
  return __builtin_shufflevector(lo, hi, 0,1,2,3,4,5,6,7,8,9,10,11,12,13,14,15);
}

static __device__ __forceinline__ v8f wmma_bf16(v16bf a, v16bf b, v8f c) {
  return __builtin_amdgcn_wmma_f32_16x16x32_bf16(
      /*neg_a=*/false, a, /*neg_b=*/false, b,
      /*c_mod=*/(short)0, c, /*reuse_a=*/false, /*reuse_b=*/false);
}

// Async copy 16 bytes global -> LDS, tracked by ASYNCcnt (CDNA5 TDM-adjacent path).
static __device__ __forceinline__ void async_copy_b128(void* lds_ptr, const void* gptr) {
  unsigned lds = (unsigned)(uintptr_t)lds_ptr;                 // LDS byte offset
  unsigned long long ga = (unsigned long long)(uintptr_t)gptr; // 64-bit global addr
  asm volatile("global_load_async_to_lds_b128 %0, %1, off"
               :: "v"(lds), "v"(ga) : "memory");
}
static __device__ __forceinline__ void wait_async0() {
  asm volatile("s_wait_asynccnt 0x0" ::: "memory");
}

// C[m,n] = sum_k A[m,k] * W[n,k] + bias[n]
// A: MTOT x 1024 (f32 or bf16), W: 1024 x 1024 f32 row-major, C: MTOT x 1024
// Block: 128 threads (4 waves), C-tile 64x64, K stepped by 32.
template<bool A_BF16, bool OUT_BF16>
__global__ __launch_bounds__(128)
void gemm_xwT(const void* __restrict__ Av, const float* __restrict__ W,
              const float* __restrict__ bias, void* __restrict__ outv)
{
  __shared__ __bf16 As[64][40];  // 64 rows x 32 K (+pad), bf16
  __shared__ __bf16 Ws[64][40];

  const int t  = threadIdx.x;
  const int wv = t >> 5;          // wave id 0..3
  const int ln = t & 31;          // lane 0..31
  const int g  = ln >> 4;         // lane half-group
  const int lr = ln & 15;         // lane row/col within 16
  const int m0 = blockIdx.y * 64;
  const int n0 = blockIdx.x * 64;

  v8f acc[4];
  #pragma unroll
  for (int j = 0; j < 4; j++) acc[j] = (v8f){0.f,0.f,0.f,0.f,0.f,0.f,0.f,0.f};

  for (int k0 = 0; k0 < HDIM; k0 += 32) {
    __syncthreads();
    // ---- stage A tile 64x32 into LDS as bf16 ----
    if (A_BF16) {
      // A already bf16: pure copy -> async global->LDS, no VGPR round trip.
      const __bf16* A = (const __bf16*)Av;
      #pragma unroll
      for (int j = 0; j < 2; j++) {
        int c8 = t + j * 128;                 // 256 chunks of 8 halves
        int r  = c8 >> 2;
        int cc = (c8 & 3) * 8;
        async_copy_b128(&As[r][cc], A + (size_t)(m0 + r) * HDIM + k0 + cc);
      }
    } else {
      const float* A = (const float*)Av;
      #pragma unroll
      for (int j = 0; j < 4; j++) {
        int f  = t + j * 128;                 // 512 float4 units
        int r  = f >> 3;
        int c4 = (f & 7) * 4;
        const float4 q4 = *(const float4*)(A + (size_t)(m0 + r) * HDIM + k0 + c4);
        As[r][c4+0] = (__bf16)q4.x; As[r][c4+1] = (__bf16)q4.y;
        As[r][c4+2] = (__bf16)q4.z; As[r][c4+3] = (__bf16)q4.w;
      }
    }
    // ---- stage W tile 64x32 (rows n0..n0+63), f32 -> bf16 ----
    #pragma unroll
    for (int j = 0; j < 4; j++) {
      int f  = t + j * 128;
      int r  = f >> 3;
      int c4 = (f & 7) * 4;
      const float4 q4 = *(const float4*)(W + (size_t)(n0 + r) * HDIM + k0 + c4);
      Ws[r][c4+0] = (__bf16)q4.x; Ws[r][c4+1] = (__bf16)q4.y;
      Ws[r][c4+2] = (__bf16)q4.z; Ws[r][c4+3] = (__bf16)q4.w;
    }
    if (A_BF16) wait_async0();
    __syncthreads();

    // A fragment: rows wv*16..wv*16+15; lane holds row lr, K chunks at 8g / 16+8g
    v8bf alo = *(const v8bf*)&As[wv*16 + lr][8*g];
    v8bf ahi = *(const v8bf*)&As[wv*16 + lr][16 + 8*g];
    v16bf afrag = cat8(alo, ahi);

    #pragma unroll
    for (int j = 0; j < 4; j++) {
      // B fragment: col = j*16+lr (row of W), K chunks at 8g / 16+8g
      v8bf blo = *(const v8bf*)&Ws[j*16 + lr][8*g];
      v8bf bhi = *(const v8bf*)&Ws[j*16 + lr][16 + 8*g];
      acc[j] = wmma_bf16(afrag, cat8(blo, bhi), acc[j]);
    }
  }

  // ---- epilogue: bias add, store ----
  #pragma unroll
  for (int j = 0; j < 4; j++) {
    int col = n0 + j*16 + lr;
    float bv = bias[col];
    #pragma unroll
    for (int r = 0; r < 8; r++) {
      int row = m0 + wv*16 + r + 8*g;
      float v = acc[j][r] + bv;
      if (OUT_BF16) ((__bf16*)outv)[(size_t)row * HDIM + col] = (__bf16)v;
      else          ((float*) outv)[(size_t)row * HDIM + col] = v;
    }
  }
}

// One block per (batch, head, 16-row query tile). 256 threads = 8 waves.
__global__ __launch_bounds__(256)
void attn_kernel(const __bf16* __restrict__ qp, const __bf16* __restrict__ kp,
                 const __bf16* __restrict__ vp, const float* __restrict__ interv,
                 float* __restrict__ attn_w, __bf16* __restrict__ ao)
{
  __shared__ float S[16][NSEQ];     // 64 KB score strip (LDS/WGP = 320 KB)
  __shared__ float Red[4][16][16];  // 4 KB split-K reduction scratch

  const int t  = threadIdx.x;
  const int wv = t >> 5;
  const int ln = t & 31;
  const int g  = ln >> 4;
  const int lr = ln & 15;

  const int qt = blockIdx.x & 63;
  const int hd = (blockIdx.x >> 6) & 15;
  const int b  = blockIdx.x >> 10;
  const int q0 = qt * 16;
  const float NEG_INF = -__builtin_inff();

  // ---- Phase 1: scores tile-by-tile over keys (waves strided) ----
  const __bf16* qbase = qp + (size_t)(b*NSEQ + q0 + lr) * HDIM + hd*HD;
  v16bf qa0 = cat8(*(const v8bf*)(qbase +      8*g), *(const v8bf*)(qbase + 16 + 8*g));
  v16bf qa1 = cat8(*(const v8bf*)(qbase + 32 + 8*g), *(const v8bf*)(qbase + 48 + 8*g));
  const float scaling = 0.125f;  // HD^-0.5

  for (int kt = wv; kt < 64; kt += 8) {
    if (kt > qt) {  // fully causally masked tile
      #pragma unroll
      for (int r = 0; r < 8; r++) S[r + 8*g][kt*16 + lr] = NEG_INF;
      continue;
    }
    const __bf16* kbase = kp + (size_t)(b*NSEQ + kt*16 + lr) * HDIM + hd*HD;
    if (kt + 8 <= qt)  // prefetch next k-tile rows for this wave
      __builtin_prefetch(kbase + (size_t)128 * HDIM, 0, 3);
    v16bf kb0 = cat8(*(const v8bf*)(kbase + 8*g),      *(const v8bf*)(kbase + 16 + 8*g));
    v16bf kb1 = cat8(*(const v8bf*)(kbase + 32 + 8*g), *(const v8bf*)(kbase + 48 + 8*g));
    v8f s = (v8f){0.f,0.f,0.f,0.f,0.f,0.f,0.f,0.f};
    s = wmma_bf16(qa0, kb0, s);
    s = wmma_bf16(qa1, kb1, s);
    #pragma unroll
    for (int r = 0; r < 8; r++) {
      int row = r + 8*g;
      int qg  = q0 + row;
      int kg  = kt*16 + lr;
      // Unconditional load (always in-bounds) + select: no exec branching.
      float iv = interv[((size_t)b*NSEQ + qg) * NSEQ + kg];
      float v  = s[r] * scaling * iv;
      S[row][kg] = (kg <= qg) ? v : NEG_INF;
    }
  }
  __syncthreads();

  // ---- Phase 2: row softmax (2 rows per wave), write attn weights ----
  float* attw_base = attn_w + (((size_t)(b*NHEAD + hd)) * NSEQ + q0) * NSEQ;
  #pragma unroll
  for (int rr = 0; rr < 2; rr++) {
    int row = wv*2 + rr;
    float m = NEG_INF;
    for (int c = ln; c < NSEQ; c += 32) m = fmaxf(m, S[row][c]);
    #pragma unroll
    for (int off = 16; off >= 1; off >>= 1) m = fmaxf(m, __shfl_xor(m, off, 32));
    float sum = 0.f;
    for (int c = ln; c < NSEQ; c += 32) {
      float e = __expf(S[row][c] - m);   // exp(-inf)=0 handles masked cols
      S[row][c] = e;
      sum += e;
    }
    #pragma unroll
    for (int off = 16; off >= 1; off >>= 1) sum += __shfl_xor(sum, off, 32);
    float inv = 1.f / sum;
    for (int c = ln; c < NSEQ; c += 32) {
      float p = S[row][c] * inv;
      S[row][c] = p;
      attw_base[(size_t)row * NSEQ + c] = p;
    }
  }
  __syncthreads();

  // ---- Phase 3: out_tile = P(16x1024) @ V(1024x64), split-K over 8 waves ----
  const int j  = wv & 3;    // N-tile
  const int kh = wv >> 2;   // K half
  const int nkt  = (q0 + 15) / 32 + 1;   // causally needed K32 tiles
  const int mid  = (nkt + 1) >> 1;
  const int kbeg = kh ? mid : 0;
  const int kend = kh ? nkt : mid;

  v8f acc = (v8f){0.f,0.f,0.f,0.f,0.f,0.f,0.f,0.f};
  for (int kt2 = kbeg; kt2 < kend; kt2++) {
    int k0 = kt2 * 32;
    v16bf pa;   // A-frag of P, requantized bf16 from LDS f32
    #pragma unroll
    for (int hh = 0; hh < 16; hh++) {
      int kk = k0 + ((hh & 8) ? 16 : 0) + (hh & 7) + 8*g;
      pa[hh] = (__bf16)S[lr][kk];
    }
    v16bf vb;   // B-frag of V: column j*16+lr, K chunks 8g / 16+8g
    #pragma unroll
    for (int hh = 0; hh < 16; hh++) {
      int vrow = k0 + ((hh & 8) ? 16 : 0) + (hh & 7) + 8*g;
      vb[hh] = vp[(size_t)(b*NSEQ + vrow) * HDIM + hd*HD + j*16 + lr];
    }
    acc = wmma_bf16(pa, vb, acc);
  }
  __syncthreads();   // everyone done reading S
  if (kh == 1) {
    #pragma unroll
    for (int r = 0; r < 8; r++) Red[j][r + 8*g][lr] = acc[r];
  }
  __syncthreads();
  if (kh == 0) {
    #pragma unroll
    for (int r = 0; r < 8; r++) {
      float v = acc[r] + Red[j][r + 8*g][lr];
      int row = q0 + r + 8*g;   // head-interleaved store == transpose+reshape
      ao[(size_t)(b*NSEQ + row) * HDIM + hd*HD + j*16 + lr] = (__bf16)v;
    }
  }
}

extern "C" void kernel_launch(void* const* d_in, const int* in_sizes, int n_in,
                              void* d_out, int out_size, void* d_ws, size_t ws_size,
                              hipStream_t stream) {
  const float* query = (const float*)d_in[0];
  const float* key_  = (const float*)d_in[1];
  const float* value = (const float*)d_in[2];
  // d_in[3] = mask (int32 tril) -- causality computed analytically
  const float* interv = (const float*)d_in[4];
  const float* Wq = (const float*)d_in[5];
  const float* bq = (const float*)d_in[6];
  const float* Wk = (const float*)d_in[7];
  const float* bk = (const float*)d_in[8];
  const float* Wv = (const float*)d_in[9];
  const float* bv = (const float*)d_in[10];
  const float* Wo = (const float*)d_in[11];
  const float* bo = (const float*)d_in[12];

  float* out    = (float*)d_out;                          // [B,N,H]
  float* attn_w = out + (size_t)BATCH * NSEQ * HDIM;      // [B,NH,N,N]

  // Workspace: bf16 q/k/v projections + bf16 attn_out (4 x 8 MB = 32 MB)
  __bf16* qp = (__bf16*)d_ws;
  __bf16* kp = qp + (size_t)MTOT * HDIM;
  __bf16* vp = kp + (size_t)MTOT * HDIM;
  __bf16* ao = vp + (size_t)MTOT * HDIM;

  dim3 ggrid(HDIM / 64, MTOT / 64);   // (16, 64)
  gemm_xwT<false, true><<<ggrid, 128, 0, stream>>>(query, Wq, bq, qp);
  gemm_xwT<false, true><<<ggrid, 128, 0, stream>>>(key_,  Wk, bk, kp);
  gemm_xwT<false, true><<<ggrid, 128, 0, stream>>>(value, Wv, bv, vp);

  attn_kernel<<<BATCH * NHEAD * (NSEQ / 16), 256, 0, stream>>>(
      qp, kp, vp, interv, attn_w, ao);

  gemm_xwT<true, false><<<ggrid, 128, 0, stream>>>(ao, Wo, bo, out);
}
